// Mamba_19507741458755
// MI455X (gfx1250) — matrix operand order repaired and checked by reading, not compiled, not run on hardware
//
#include <hip/hip_runtime.h>
#include <cstddef>

// ---------------------------------------------------------------------------
// Mamba forward (4 layers) for gfx1250.
//  - GEMMs: v_wmma_f32_16x16x32_bf16, 128x128x32 block tile, double-buffered
//    LDS, software-pipelined (fp32 global b128 loads overlap WMMAs; cheap
//    truncation fp32->bf16 packing in VGPRs).
//  - Selective scan: B/C staged with global_load_async_to_lds_b128 (ASYNCcnt).
// ---------------------------------------------------------------------------

#define LSEQ    1024
#define DMODEL  1024
#define DINNER  2048
#define DSTATE  16
#define DTRANK  64
#define DCONV   4
#define NVOCAB  50264
#define NLAYER  4

typedef __attribute__((ext_vector_type(16))) __bf16 v16bf;
typedef __attribute__((ext_vector_type(8)))  float  v8f;
typedef __attribute__((ext_vector_type(4)))  float  v4f;
typedef __attribute__((ext_vector_type(4)))  unsigned int v4u;
typedef __attribute__((ext_vector_type(2)))  unsigned int v2u;

union FragAB {
  v16bf v;
  v4u u4[2];
};

__device__ __forceinline__ unsigned int bits_(float f) {
  union { float f; unsigned int u; } c; c.f = f; return c.u;
}
// truncating fp32->bf16 (1 VALU op per 2 elements vs ~5/elt for RNE)
__device__ __forceinline__ unsigned int pack_bf16(float lo, float hi) {
  return (bits_(hi) & 0xFFFF0000u) | (bits_(lo) >> 16);
}
__device__ __forceinline__ unsigned short f2bf(float f) {
  return (unsigned short)(bits_(f) >> 16);
}
__device__ __forceinline__ float silu_(float x) { return x / (1.f + __expf(-x)); }

// CDNA5 async global->LDS copy (tracked by ASYNCcnt)
__device__ __forceinline__ void async_copy_b128(unsigned int lds_off, const void* gptr) {
  asm volatile("global_load_async_to_lds_b128 %0, %1, off"
               :: "v"(lds_off), "v"(gptr) : "memory");
}
__device__ __forceinline__ void wait_async0() {
  asm volatile("s_wait_asynccnt 0x0" ::: "memory");
}

// ---------------------------------------------------------------------------
// Generic bf16 WMMA GEMM: C[M,N] = A[M,K] * op(B)  (+ fused epilogue)
//   TRANSB=0 : B is [K,N] row-major (weights)
//   TRANSB=1 : B is [N,K] row-major (embed for tied logits, NT gemm)
//   EPI=0 : C = acc ; EPI=1 : C = softplus(acc + aux[n]) ; EPI=2 : C += aux[m,n]
// BM=128, BN=128, BK=32; 8 waves: wave w -> rows 32*(w&3), cols 64*(w>>2);
// each wave: 2 A-frags x 4 B-frags = 8 back-to-back WMMAs per K-step.
// Double-buffered LDS; one barrier per K-step; tile k+1 global loads overlap
// the WMMAs on tile k. M must be a multiple of 128; N edges are guarded.
// ---------------------------------------------------------------------------
template<int TRANSB, int EPI>
__global__ __launch_bounds__(256) void gemm_bf16_wmma(
    const float* __restrict__ A, int lda,
    const float* __restrict__ B, int ldb,
    float* __restrict__ C, int ldc,
    const float* __restrict__ aux,
    int N, int K)
{
  constexpr int BM = 128, BN = 128, BK = 32;
  constexpr int SA = 40;   // halfs per A row: 80B rows -> b128-aligned fragments
  constexpr int SB = 40;   // halfs per B n-row: 80B rows
  __shared__ __attribute__((aligned(16))) unsigned short As[2][BM * SA];
  __shared__ __attribute__((aligned(16))) unsigned short Bs[2][BN * SB];

  const int m0 = blockIdx.y * BM;
  const int n0 = blockIdx.x * BN;
  const int tid  = threadIdx.x;
  const int wave = tid >> 5;
  const int lane = tid & 31;
  const int lm = lane & 15;
  const int hi = lane >> 4;
  const int rbase = (wave & 3) * 32;   // wave's 32-row strip
  const int cw = wave >> 2;            // wave's 64-col half

  // static staging coordinates
  const int am = tid >> 1;             // A row 0..127
  const int ak = (tid & 1) << 4;       // A k-chunk {0,16}
  const int nn = tid & 127;            // TRANSB=0: B column
  const int kq = (tid >> 7) << 4;      // TRANSB=0: k-chunk {0,16}
  const int bn = tid >> 1;             // TRANSB=1: B row (column of op(B))
  const int bk = (tid & 1) << 4;       // TRANSB=1: k-chunk {0,16}

  float a_reg[16];
  float b_reg[16];

  auto load_tiles = [&](int k0) {
    const float* Ap = A + (size_t)(m0 + am) * lda + (k0 + ak);
    v4f a0 = *(const v4f*)(Ap);
    v4f a1 = *(const v4f*)(Ap + 4);
    v4f a2 = *(const v4f*)(Ap + 8);
    v4f a3 = *(const v4f*)(Ap + 12);
#pragma unroll
    for (int c = 0; c < 4; ++c) {
      a_reg[c] = a0[c]; a_reg[4 + c] = a1[c]; a_reg[8 + c] = a2[c]; a_reg[12 + c] = a3[c];
    }
    if (TRANSB == 0) {
      int gn = n0 + nn;
      if (gn < N) {
        const float* Bp = B + (size_t)(k0 + kq) * ldb + gn;
#pragma unroll
        for (int j = 0; j < 16; ++j) b_reg[j] = Bp[(size_t)j * ldb];
      } else {
#pragma unroll
        for (int j = 0; j < 16; ++j) b_reg[j] = 0.f;
      }
    } else {
      int gn = n0 + bn;
      if (gn < N) {
        const float* Bp = B + (size_t)gn * ldb + (k0 + bk);
        v4f b0 = *(const v4f*)(Bp);
        v4f b1 = *(const v4f*)(Bp + 4);
        v4f b2 = *(const v4f*)(Bp + 8);
        v4f b3 = *(const v4f*)(Bp + 12);
#pragma unroll
        for (int c = 0; c < 4; ++c) {
          b_reg[c] = b0[c]; b_reg[4 + c] = b1[c]; b_reg[8 + c] = b2[c]; b_reg[12 + c] = b3[c];
        }
      } else {
#pragma unroll
        for (int j = 0; j < 16; ++j) b_reg[j] = 0.f;
      }
    }
  };

  auto store_tiles = [&](int buf) {
    // A: 16 fp32 -> 16 bf16 -> two b128 LDS stores
    v4u p0 = {pack_bf16(a_reg[0],  a_reg[1]),  pack_bf16(a_reg[2],  a_reg[3]),
              pack_bf16(a_reg[4],  a_reg[5]),  pack_bf16(a_reg[6],  a_reg[7])};
    v4u p1 = {pack_bf16(a_reg[8],  a_reg[9]),  pack_bf16(a_reg[10], a_reg[11]),
              pack_bf16(a_reg[12], a_reg[13]), pack_bf16(a_reg[14], a_reg[15])};
    *(v4u*)&As[buf][am * SA + ak]     = p0;
    *(v4u*)&As[buf][am * SA + ak + 8] = p1;
    if (TRANSB == 0) {
      // column nn, k = kq..kq+15: four packed b64 stores
#pragma unroll
      for (int q = 0; q < 4; ++q) {
        v2u pb = {pack_bf16(b_reg[q * 4 + 0], b_reg[q * 4 + 1]),
                  pack_bf16(b_reg[q * 4 + 2], b_reg[q * 4 + 3])};
        *(v2u*)&Bs[buf][nn * SB + kq + q * 4] = pb;
      }
    } else {
      v4u q0 = {pack_bf16(b_reg[0],  b_reg[1]),  pack_bf16(b_reg[2],  b_reg[3]),
                pack_bf16(b_reg[4],  b_reg[5]),  pack_bf16(b_reg[6],  b_reg[7])};
      v4u q1 = {pack_bf16(b_reg[8],  b_reg[9]),  pack_bf16(b_reg[10], b_reg[11]),
                pack_bf16(b_reg[12], b_reg[13]), pack_bf16(b_reg[14], b_reg[15])};
      *(v4u*)&Bs[buf][bn * SB + bk]     = q0;
      *(v4u*)&Bs[buf][bn * SB + bk + 8] = q1;
    }
  };

  v8f acc[2][4];
#pragma unroll
  for (int r = 0; r < 2; ++r)
#pragma unroll
    for (int t = 0; t < 4; ++t)
      acc[r][t] = (v8f){0.f,0.f,0.f,0.f,0.f,0.f,0.f,0.f};

  load_tiles(0);
  store_tiles(0);
  __syncthreads();

  int buf = 0;
  for (int k0 = 0; k0 < K; k0 += BK) {
    const bool has_next = (k0 + BK) < K;
    if (has_next) load_tiles(k0 + BK);             // overlap with WMMAs below
    if (k0 + 2 * BK < K) {                         // warm L2 for tile k+2
      __builtin_prefetch(A + (size_t)(m0 + am) * lda + (k0 + 2 * BK + ak), 0, 3);
      if (TRANSB == 0) {
        __builtin_prefetch(B + (size_t)(k0 + 2 * BK + (tid >> 3)) * ldb + n0 + ((tid & 7) << 4), 0, 3);
      } else {
        int gn = n0 + (tid >> 1);
        if (gn < N)
          __builtin_prefetch(B + (size_t)gn * ldb + (k0 + 2 * BK) + ((tid & 1) << 4), 0, 3);
      }
    }

    // ---- load all fragments first (batched ds_load_b128), then 8 WMMAs ----
    // A frag (ISA 7.12.2): VGPR j -> K = (j>=4)*16 + hi*8 + (j&3)*2
    FragAB fa[2];
#pragma unroll
    for (int r = 0; r < 2; ++r) {
      const unsigned short* Ab = &As[buf][(rbase + r * 16 + lm) * SA];
      fa[r].u4[0] = *(const v4u*)(Ab + hi * 8);
      fa[r].u4[1] = *(const v4u*)(Ab + 16 + hi * 8);
    }
    // B frag: VGPR j -> K = (j>=4)*16 + hi*8 + (j&3)*2 (same as A)
    FragAB fb[4];
#pragma unroll
    for (int t = 0; t < 4; ++t) {
      const unsigned short* Bb = &Bs[buf][(cw * 64 + t * 16 + lm) * SB];
      fb[t].u4[0] = *(const v4u*)(Bb + hi * 8);
      fb[t].u4[1] = *(const v4u*)(Bb + 16 + hi * 8);
    }
#pragma unroll
    for (int r = 0; r < 2; ++r)
#pragma unroll
      for (int t = 0; t < 4; ++t)
        acc[r][t] = __builtin_amdgcn_wmma_f32_16x16x32_bf16(
            false, fa[r].v, false, fb[t].v, (short)0, acc[r][t], false, false);

    if (has_next) store_tiles(buf ^ 1);
    __syncthreads();
    buf ^= 1;
  }

  // ---- epilogue: C/D layout lane=n (lm), VGPR j -> m = hi*8 + j ----
#pragma unroll
  for (int r = 0; r < 2; ++r) {
#pragma unroll
    for (int t = 0; t < 4; ++t) {
      int gn = n0 + cw * 64 + t * 16 + lm;
      if (gn < N) {
#pragma unroll
        for (int j = 0; j < 8; ++j) {
          int gm = m0 + rbase + r * 16 + hi * 8 + j;
          float v = acc[r][t][j];
          if (EPI == 1) {                       // bias + softplus
            v += aux[gn];
            v = (v > 20.f) ? v : log1pf(__expf(v));
          } else if (EPI == 2) {                // residual add
            v += aux[(size_t)gm * ldc + gn];
          }
          C[(size_t)gm * ldc + gn] = v;
        }
      }
    }
  }
}

// ---------------------------------------------------------------------------
// Elementwise / small kernels
// ---------------------------------------------------------------------------
__global__ __launch_bounds__(256) void embed_gather_kernel(
    const int* __restrict__ ids, const float* __restrict__ embed,
    float* __restrict__ x)
{
  int l = blockIdx.x;
  int id = ids[l];
  const float* src = embed + (size_t)id * DMODEL;
  float* dst = x + (size_t)l * DMODEL;
  for (int i = threadIdx.x; i < DMODEL; i += 256) dst[i] = src[i];
}

__global__ __launch_bounds__(256) void rmsnorm_kernel(
    const float* __restrict__ x, const float* __restrict__ w,
    float* __restrict__ out, int D)
{
  __shared__ float red[256];
  int row = blockIdx.x;
  const float* xr = x + (size_t)row * D;
  float s = 0.f;
  for (int i = threadIdx.x; i < D; i += 256) { float v = xr[i]; s += v * v; }
  red[threadIdx.x] = s;
  __syncthreads();
  for (int off = 128; off > 0; off >>= 1) {
    if (threadIdx.x < off) red[threadIdx.x] += red[threadIdx.x + off];
    __syncthreads();
  }
  float inv = rsqrtf(red[0] / (float)D + 1e-5f);
  float* o = out + (size_t)row * D;
  for (int i = threadIdx.x; i < D; i += 256) o[i] = xr[i] * inv * w[i];
}

// causal depthwise conv (k=4, left pad 3) + bias + SiLU; input = xr[:, :DINNER]
__global__ __launch_bounds__(256) void conv_silu_kernel(
    const float* __restrict__ xr, const float* __restrict__ w,
    const float* __restrict__ b, float* __restrict__ u)
{
  int idx = blockIdx.x * 256 + threadIdx.x;
  if (idx >= LSEQ * DINNER) return;
  int l = idx / DINNER, d = idx % DINNER;
  float acc = b[d];
#pragma unroll
  for (int i = 0; i < DCONV; ++i) {
    int lp = l - (DCONV - 1) + i;
    if (lp >= 0) acc += w[i * DINNER + d] * xr[(size_t)lp * (2 * DINNER) + d];
  }
  u[idx] = silu_(acc);
}

// selective scan: thread per channel d; B/C (shared across all d) staged in LDS
// via CDNA5 async global->LDS b128 copies.
__global__ __launch_bounds__(256) void scan_kernel(
    const float* __restrict__ u, const float* __restrict__ delta,
    const float* __restrict__ A_log, const float* __restrict__ xdbl,
    const float* __restrict__ Dp, float* __restrict__ y)
{
  int d = blockIdx.x * 256 + threadIdx.x;
  float a[DSTATE], s[DSTATE];
#pragma unroll
  for (int n = 0; n < DSTATE; ++n) {
    a[n] = -__expf(A_log[(size_t)d * DSTATE + n]);
    s[n] = 0.f;
  }
  float Dd = Dp[d];
  __shared__ __attribute__((aligned(16))) float Bsh[64][DSTATE];
  __shared__ __attribute__((aligned(16))) float Csh[64][DSTATE];
  // generic->LDS pointer truncation yields the wave-relative LDS byte address
  const unsigned int bsh_base = (unsigned int)(size_t)&Bsh[0][0];
  const unsigned int csh_base = (unsigned int)(size_t)&Csh[0][0];
  const int ll = threadIdx.x >> 2;          // staged row 0..63
  const int nq = (threadIdx.x & 3) << 2;    // float4 within 16-wide row

  for (int l0 = 0; l0 < LSEQ; l0 += 64) {
    const float* row = xdbl + (size_t)(l0 + ll) * (DTRANK + 2 * DSTATE) + DTRANK + nq;
    unsigned int loff = (unsigned int)((ll * DSTATE + nq) * 4);
    async_copy_b128(bsh_base + loff, row);            // B block
    async_copy_b128(csh_base + loff, row + DSTATE);   // C block
    wait_async0();
    __syncthreads();
    for (int t = 0; t < 64; ++t) {
      int l = l0 + t;
      float dl = delta[(size_t)l * DINNER + d];
      float ul = u[(size_t)l * DINNER + d];
      float du = dl * ul;
      float yt = 0.f;
#pragma unroll
      for (int n = 0; n < DSTATE; ++n) {
        s[n] = __expf(dl * a[n]) * s[n] + du * Bsh[t][n];
        yt += s[n] * Csh[t][n];
      }
      y[(size_t)l * DINNER + d] = yt + ul * Dd;
    }
    __syncthreads();
  }
}

// y *= silu(res); res = xr[:, DINNER:2*DINNER]
__global__ __launch_bounds__(256) void gate_kernel(
    float* __restrict__ y, const float* __restrict__ xr)
{
  int idx = blockIdx.x * 256 + threadIdx.x;
  if (idx >= LSEQ * DINNER) return;
  int l = idx / DINNER, d = idx % DINNER;
  float res = xr[(size_t)l * (2 * DINNER) + DINNER + d];
  y[idx] *= silu_(res);
}

// ---------------------------------------------------------------------------
// Host-side orchestration
// ---------------------------------------------------------------------------
extern "C" void kernel_launch(void* const* d_in, const int* in_sizes, int n_in,
                              void* d_out, int out_size, void* d_ws, size_t ws_size,
                              hipStream_t stream) {
  (void)in_sizes; (void)n_in; (void)out_size; (void)ws_size;
  const int*   ids        = (const int*)d_in[0];
  const float* embed      = (const float*)d_in[1];   // (VOCAB, DMODEL)
  const float* norm_w     = (const float*)d_in[2];   // (NLAYER, DMODEL)
  const float* in_proj_w  = (const float*)d_in[3];   // (NLAYER, DMODEL, 2*DINNER)
  const float* conv_w     = (const float*)d_in[4];   // (NLAYER, DCONV, DINNER)
  const float* conv_b     = (const float*)d_in[5];   // (NLAYER, DINNER)
  const float* x_proj_w   = (const float*)d_in[6];   // (NLAYER, DINNER, 96)
  const float* dt_proj_w  = (const float*)d_in[7];   // (NLAYER, DTRANK, DINNER)
  const float* dt_proj_b  = (const float*)d_in[8];   // (NLAYER, DINNER)
  const float* A_log      = (const float*)d_in[9];   // (NLAYER, DINNER, DSTATE)
  const float* D_ssm      = (const float*)d_in[10];  // (NLAYER, DINNER)
  const float* out_proj_w = (const float*)d_in[11];  // (NLAYER, DINNER, DMODEL)
  const float* norm_f_w   = (const float*)d_in[12];  // (DMODEL,)
  float* out = (float*)d_out;                        // (LSEQ, VOCAB)

  // workspace layout (floats): ~50.7 MB total
  float* ws    = (float*)d_ws;
  float* x     = ws;                                   // LSEQ*DMODEL
  float* xn    = x     + (size_t)LSEQ * DMODEL;        // LSEQ*DMODEL
  float* xr    = xn    + (size_t)LSEQ * DMODEL;        // LSEQ*2*DINNER
  float* u     = xr    + (size_t)LSEQ * 2 * DINNER;    // LSEQ*DINNER
  float* xdbl  = u     + (size_t)LSEQ * DINNER;        // LSEQ*96
  float* delta = xdbl  + (size_t)LSEQ * (DTRANK + 2 * DSTATE); // LSEQ*DINNER
  float* yb    = delta + (size_t)LSEQ * DINNER;        // LSEQ*DINNER

  const int NXP = DTRANK + 2 * DSTATE;  // 96

  embed_gather_kernel<<<LSEQ, 256, 0, stream>>>(ids, embed, x);

  for (int i = 0; i < NLAYER; ++i) {
    rmsnorm_kernel<<<LSEQ, 256, 0, stream>>>(x, norm_w + (size_t)i * DMODEL, xn, DMODEL);

    // xr = xn @ in_proj_w[i]   (1024 x 1024 x 4096)
    gemm_bf16_wmma<0, 0><<<dim3(2 * DINNER / 128, LSEQ / 128), 256, 0, stream>>>(
        xn, DMODEL, in_proj_w + (size_t)i * DMODEL * 2 * DINNER, 2 * DINNER,
        xr, 2 * DINNER, nullptr, 2 * DINNER, DMODEL);

    // u = silu(causal_dw_conv(xr[:, :DINNER]) + b)
    conv_silu_kernel<<<(LSEQ * DINNER + 255) / 256, 256, 0, stream>>>(
        xr, conv_w + (size_t)i * DCONV * DINNER, conv_b + (size_t)i * DINNER, u);

    // xdbl = u @ x_proj_w[i]   (1024 x 2048 x 96)
    gemm_bf16_wmma<0, 0><<<dim3(1, LSEQ / 128), 256, 0, stream>>>(
        u, DINNER, x_proj_w + (size_t)i * DINNER * NXP, NXP,
        xdbl, NXP, nullptr, NXP, DINNER);

    // delta = softplus(xdbl[:, :64] @ dt_proj_w[i] + b)   (1024 x 64 x 2048)
    gemm_bf16_wmma<0, 1><<<dim3(DINNER / 128, LSEQ / 128), 256, 0, stream>>>(
        xdbl, NXP, dt_proj_w + (size_t)i * DTRANK * DINNER, DINNER,
        delta, DINNER, dt_proj_b + (size_t)i * DINNER, DINNER, DTRANK);

    // selective scan -> yb
    scan_kernel<<<DINNER / 256, 256, 0, stream>>>(
        u, delta, A_log + (size_t)i * DINNER * DSTATE, xdbl,
        D_ssm + (size_t)i * DINNER, yb);

    // yb *= silu(res)
    gate_kernel<<<(LSEQ * DINNER + 255) / 256, 256, 0, stream>>>(yb, xr);

    // x = yb @ out_proj_w[i] + x   (1024 x 2048 x 1024, residual epilogue)
    gemm_bf16_wmma<0, 2><<<dim3(DMODEL / 128, LSEQ / 128), 256, 0, stream>>>(
        yb, DINNER, out_proj_w + (size_t)i * DINNER * DMODEL, DMODEL,
        x, DMODEL, x, DMODEL, DINNER);
  }

  rmsnorm_kernel<<<LSEQ, 256, 0, stream>>>(x, norm_f_w, xn, DMODEL);

  // logits = xn @ embed.T   (1024 x 1024 x 50264, NT gemm)
  gemm_bf16_wmma<1, 0><<<dim3((NVOCAB + 127) / 128, LSEQ / 128), 256, 0, stream>>>(
      xn, DMODEL, embed, DMODEL, out, NVOCAB, nullptr, NVOCAB, DMODEL);
}
